// SLSTM_cell_36816459661906
// MI455X (gfx1250) — compile-verified, matrix operand order checked by reading, not run on hardware
//
#include <hip/hip_runtime.h>
#include <math.h>

// ---------------------------------------------------------------------------
// SLSTM cell for MI455X (gfx1250, wave32, WMMA 16x16x32 f16).
// 4 fused gate GEMMs (x@Wi^T + h@Wh^T), split-f16 weights for ~fp32 accuracy,
// LIF epilogue fused into the GEMM, spike-rate reduction via LDS+global atomics.
// ---------------------------------------------------------------------------

typedef __attribute__((ext_vector_type(16))) _Float16 v16h;
typedef __attribute__((ext_vector_type(8)))  _Float16 v8h;
typedef __attribute__((ext_vector_type(8)))  float    v8f;
typedef __attribute__((ext_vector_type(4)))  float    v4f;

#define BATCH 4096
#define KDIM  2048   // D_IN == D_H
#define PDIM  1024
#define DH    2048

#define BM 128       // block tile M (batch rows)
#define BN 128       // block tile N (gate units)
#define KT 32        // K per WMMA / per LDS stage
#define LSTR 40      // LDS row stride in halves (80B, 16B aligned, conflict-padded)

__device__ __forceinline__ float clip01(float b) {
    return fminf(fmaxf(b, 0.0f), 1.0f);
}

// ---------------------------------------------------------------------------
// Zero the spike row-sum accumulators in workspace.
// ---------------------------------------------------------------------------
__global__ void zero_ws(float* __restrict__ p, int n) {
    int i = blockIdx.x * blockDim.x + threadIdx.x;
    if (i < n) p[i] = 0.0f;
}

// ---------------------------------------------------------------------------
// One gate:  enc = X @ Wi^T + bi + H @ Wh^T + bh
//            mem_new = clip(beta)*mem + enc - (mem>thr)*thr
//            spk     = (mem_new - thr) > 0
// writes mem_new to mem_out, accumulates per-row sum(spk) into rowsum[4096].
// ---------------------------------------------------------------------------
__global__ __launch_bounds__(256)
void gate_gemm(const float* __restrict__ X,      // [4096 x 2048] binary spikes
               const float* __restrict__ H,      // [4096 x 2048] binary spikes
               const float* __restrict__ Wi,     // [1024 x 2048]
               const float* __restrict__ Wh,     // [1024 x 2048]
               const float* __restrict__ bi,     // [1024]
               const float* __restrict__ bh,     // [1024]
               const float* __restrict__ mem_in, // [4096 x 1024]
               const float* __restrict__ beta_p,
               const float* __restrict__ thr_p,
               float* __restrict__ mem_out,      // [4096 x 1024]
               float* __restrict__ rowsum)       // [4096]
{
    __shared__ _Float16 lA [BM * LSTR];
    __shared__ _Float16 lBh[BN * LSTR];
    __shared__ _Float16 lBl[BN * LSTR];
    __shared__ float    lrow[BM];

    const int tid  = threadIdx.x;
    const int lane = tid & 31;
    const int wave = tid >> 5;
    const int mw   = wave & 1;        // 2 wave rows  (64 batch rows each)
    const int nw   = wave >> 1;       // 4 wave cols  (32 units each)
    const int blockM = blockIdx.x * BM;
    const int blockN = blockIdx.y * BN;

    if (tid < BM) lrow[tid] = 0.0f;

    // cooperative loader mapping: 2 threads per row, 16 floats each
    const int lr = tid >> 1;
    const int lk = (tid & 1) * 16;

    // fragment mapping (ISA 7.12.2): lane -> row = lane%16, k0 = lane<16 ? 0 : 8
    const int fm = lane & 15;
    const int k0 = (lane < 16) ? 0 : 8;

    v8f acc[4][2];
#pragma unroll
    for (int t = 0; t < 4; ++t)
#pragma unroll
        for (int u = 0; u < 2; ++u) acc[t][u] = (v8f)(0.0f);

#pragma unroll 1
    for (int phase = 0; phase < 2; ++phase) {
        const float* __restrict__ Asrc = phase ? H  : X;
        const float* __restrict__ Wsrc = phase ? Wh : Wi;

#pragma unroll 1
        for (int kk = 0; kk < KDIM; kk += KT) {
            __syncthreads();   // protect previous stage's LDS reads

            const float* ap = Asrc + (size_t)(blockM + lr) * KDIM + kk + lk;
            const float* wp = Wsrc + (size_t)(blockN + lr) * KDIM + kk + lk;

            if (kk + KT < KDIM) {                 // gfx1250 global_prefetch_b8
                __builtin_prefetch(ap + KT, 0, 3);
                __builtin_prefetch(wp + KT, 0, 3);
            }

            v4f a0 = *(const v4f*)(ap);
            v4f a1 = *(const v4f*)(ap + 4);
            v4f a2 = *(const v4f*)(ap + 8);
            v4f a3 = *(const v4f*)(ap + 12);
            v4f w0 = *(const v4f*)(wp);
            v4f w1 = *(const v4f*)(wp + 4);
            v4f w2 = *(const v4f*)(wp + 8);
            v4f w3 = *(const v4f*)(wp + 12);

            // A: spikes are exact in f16
            {
                v8h h0, h1;
#pragma unroll
                for (int j = 0; j < 4; ++j) {
                    h0[j]     = (_Float16)a0[j];
                    h0[j + 4] = (_Float16)a1[j];
                    h1[j]     = (_Float16)a2[j];
                    h1[j + 4] = (_Float16)a3[j];
                }
                _Float16* la = &lA[lr * LSTR + lk];
                *(v8h*)(la)     = h0;
                *(v8h*)(la + 8) = h1;
            }
            // W: split into hi(f16) + lo(f16) for ~fp32 fidelity
            {
                v8h hh0, hh1, hl0, hl1;
#pragma unroll
                for (int j = 0; j < 4; ++j) {
                    float x;
                    _Float16 hi;
                    x = w0[j]; hi = (_Float16)x; hh0[j]     = hi; hl0[j]     = (_Float16)(x - (float)hi);
                    x = w1[j]; hi = (_Float16)x; hh0[j + 4] = hi; hl0[j + 4] = (_Float16)(x - (float)hi);
                    x = w2[j]; hi = (_Float16)x; hh1[j]     = hi; hl1[j]     = (_Float16)(x - (float)hi);
                    x = w3[j]; hi = (_Float16)x; hh1[j + 4] = hi; hl1[j + 4] = (_Float16)(x - (float)hi);
                }
                _Float16* lh = &lBh[lr * LSTR + lk];
                _Float16* ll = &lBl[lr * LSTR + lk];
                *(v8h*)(lh)     = hh0;  *(v8h*)(lh + 8) = hh1;
                *(v8h*)(ll)     = hl0;  *(v8h*)(ll + 8) = hl1;
            }
            __syncthreads();

            // gather fragments (two b128 LDS reads each: k0..k0+7, k0+16..k0+23)
            v16h afr[4];
#pragma unroll
            for (int t = 0; t < 4; ++t) {
                const _Float16* p = &lA[(mw * 64 + t * 16 + fm) * LSTR + k0];
                v8h lo8 = *(const v8h*)(p);
                v8h hi8 = *(const v8h*)(p + 16);
                v16h f;
#pragma unroll
                for (int j = 0; j < 8; ++j) { f[j] = lo8[j]; f[j + 8] = hi8[j]; }
                afr[t] = f;
            }
            v16h bhi[2], blo[2];
#pragma unroll
            for (int u = 0; u < 2; ++u) {
                const _Float16* ph = &lBh[(nw * 32 + u * 16 + fm) * LSTR + k0];
                const _Float16* pl = &lBl[(nw * 32 + u * 16 + fm) * LSTR + k0];
                v8h a = *(const v8h*)(ph);
                v8h b = *(const v8h*)(ph + 16);
                v8h c = *(const v8h*)(pl);
                v8h d = *(const v8h*)(pl + 16);
                v16h fh, fl;
#pragma unroll
                for (int j = 0; j < 8; ++j) {
                    fh[j] = a[j]; fh[j + 8] = b[j];
                    fl[j] = c[j]; fl[j + 8] = d[j];
                }
                bhi[u] = fh; blo[u] = fl;
            }

#pragma unroll
            for (int t = 0; t < 4; ++t)
#pragma unroll
                for (int u = 0; u < 2; ++u) {
                    acc[t][u] = __builtin_amdgcn_wmma_f32_16x16x32_f16(
                        false, afr[t], false, bhi[u], (short)0, acc[t][u], false, false);
                    acc[t][u] = __builtin_amdgcn_wmma_f32_16x16x32_f16(
                        false, afr[t], false, blo[u], (short)0, acc[t][u], false, false);
                }
        }
    }

    // ---- fused LIF epilogue -------------------------------------------------
    const float beta = clip01(beta_p[0]);
    const float thr  = thr_p[0];

#pragma unroll
    for (int t = 0; t < 4; ++t)
#pragma unroll
        for (int u = 0; u < 2; ++u) {
            const int gn   = blockN + nw * 32 + u * 16 + (lane & 15);
            const float bias = bi[gn] + bh[gn];
#pragma unroll
            for (int j = 0; j < 8; ++j) {
                const int lm = mw * 64 + t * 16 + j + ((lane < 16) ? 0 : 8);
                const int gm = blockM + lm;
                const float enc   = acc[t][u][j] + bias;
                const float m     = mem_in[(size_t)gm * PDIM + gn];
                const float reset = (m - thr > 0.0f) ? 1.0f : 0.0f;
                const float mnew  = beta * m + enc - reset * thr;
                mem_out[(size_t)gm * PDIM + gn] = mnew;
                const float spk = (mnew - thr > 0.0f) ? 1.0f : 0.0f;
                atomicAdd(&lrow[lm], spk);      // ds_add_f32
            }
        }
    __syncthreads();
    if (tid < BM) atomicAdd(&rowsum[blockM + tid], lrow[tid]);
}

// ---------------------------------------------------------------------------
// Final elementwise stage: gate scalars from spike rates, syn/mem_h/hidden.
// ---------------------------------------------------------------------------
__global__ __launch_bounds__(256)
void final_elem(const float* __restrict__ syn_prev,   // [4096 x 2048]
                const float* __restrict__ mem_hidden, // [4096 x 2048]
                const float* __restrict__ rs_f,
                const float* __restrict__ rs_i,
                const float* __restrict__ rs_g,
                const float* __restrict__ rs_o,
                const float* __restrict__ beta_h_p,
                const float* __restrict__ thr_h_p,
                float* __restrict__ out_hidden,
                float* __restrict__ out_syn,
                float* __restrict__ out_memh)
{
    const size_t idx = (size_t)blockIdx.x * blockDim.x + threadIdx.x;
    if (idx >= (size_t)BATCH * DH) return;
    const int b = (int)(idx / DH);

    const float inv = 1.0f / (float)PDIM;
    const float f = 1.0f / (1.0f + expf(-rs_f[b] * inv));
    const float i = 1.0f / (1.0f + expf(-rs_i[b] * inv));
    const float g = tanhf(rs_g[b] * inv);
    const float o = 1.0f / (1.0f + expf(-rs_o[b] * inv));

    const float syn    = f * syn_prev[idx] + i * g;
    const float mh_in  = o * tanhf(syn);
    const float beta   = clip01(beta_h_p[0]);
    const float thr    = thr_h_p[0];
    const float m      = mem_hidden[idx];
    const float reset  = (m - thr > 0.0f) ? 1.0f : 0.0f;
    const float mh     = beta * m + mh_in - reset * thr;

    out_syn[idx]    = syn;
    out_memh[idx]   = mh;
    out_hidden[idx] = (mh - thr > 0.0f) ? 1.0f : 0.0f;
}

// ---------------------------------------------------------------------------
extern "C" void kernel_launch(void* const* d_in, const int* in_sizes, int n_in,
                              void* d_out, int out_size, void* d_ws, size_t ws_size,
                              hipStream_t stream) {
    (void)in_sizes; (void)n_in; (void)out_size; (void)ws_size;

    const float* spk1 = (const float*)d_in[0];
    const float* spk2 = (const float*)d_in[1];
    const float* spk3 = (const float*)d_in[2];
    const float* spk4 = (const float*)d_in[3];
    const float* hprev = (const float*)d_in[4];
    const float* syn_prev = (const float*)d_in[5];
    const float* mem_f = (const float*)d_in[6];
    const float* mem_i = (const float*)d_in[7];
    const float* mem_g = (const float*)d_in[8];
    const float* mem_o = (const float*)d_in[9];
    const float* mem_h = (const float*)d_in[10];
    const float* Wif_w = (const float*)d_in[11]; const float* Wif_b = (const float*)d_in[12];
    const float* Whf_w = (const float*)d_in[13]; const float* Whf_b = (const float*)d_in[14];
    const float* Wii_w = (const float*)d_in[15]; const float* Wii_b = (const float*)d_in[16];
    const float* Whi_w = (const float*)d_in[17]; const float* Whi_b = (const float*)d_in[18];
    const float* Wig_w = (const float*)d_in[19]; const float* Wig_b = (const float*)d_in[20];
    const float* Whg_w = (const float*)d_in[21]; const float* Whg_b = (const float*)d_in[22];
    const float* Wio_w = (const float*)d_in[23]; const float* Wio_b = (const float*)d_in[24];
    const float* Who_w = (const float*)d_in[25]; const float* Who_b = (const float*)d_in[26];
    const float* beta_f = (const float*)d_in[27]; const float* thr_f = (const float*)d_in[28];
    const float* beta_i = (const float*)d_in[29]; const float* thr_i = (const float*)d_in[30];
    const float* beta_g = (const float*)d_in[31]; const float* thr_g = (const float*)d_in[32];
    const float* beta_o = (const float*)d_in[33]; const float* thr_o = (const float*)d_in[34];
    const float* beta_h = (const float*)d_in[35]; const float* thr_h = (const float*)d_in[36];

    float* out = (float*)d_out;
    const size_t BDH = (size_t)BATCH * DH;   // 8M
    const size_t BP  = (size_t)BATCH * PDIM; // 4M
    float* o_hidden = out;
    float* o_syn    = out + BDH;
    float* o_memf   = out + 2 * BDH;
    float* o_memi   = o_memf + BP;
    float* o_memg   = o_memi + BP;
    float* o_memo   = o_memg + BP;
    float* o_memh   = o_memo + BP;

    float* rs = (float*)d_ws;                // 4 x 4096 spike row-sums
    zero_ws<<<(4 * BATCH + 255) / 256, 256, 0, stream>>>(rs, 4 * BATCH);

    dim3 grid(BATCH / BM, PDIM / BN);        // 32 x 8 blocks per gate
    gate_gemm<<<grid, 256, 0, stream>>>(spk1, hprev, Wif_w, Whf_w, Wif_b, Whf_b,
                                        mem_f, beta_f, thr_f, o_memf, rs + 0 * BATCH);
    gate_gemm<<<grid, 256, 0, stream>>>(spk2, hprev, Wii_w, Whi_w, Wii_b, Whi_b,
                                        mem_i, beta_i, thr_i, o_memi, rs + 1 * BATCH);
    gate_gemm<<<grid, 256, 0, stream>>>(spk3, hprev, Wig_w, Whg_w, Wig_b, Whg_b,
                                        mem_g, beta_g, thr_g, o_memg, rs + 2 * BATCH);
    gate_gemm<<<grid, 256, 0, stream>>>(spk4, hprev, Wio_w, Who_w, Wio_b, Who_b,
                                        mem_o, beta_o, thr_o, o_memo, rs + 3 * BATCH);

    final_elem<<<(unsigned)((BDH + 255) / 256), 256, 0, stream>>>(
        syn_prev, mem_h, rs, rs + BATCH, rs + 2 * BATCH, rs + 3 * BATCH,
        beta_h, thr_h, o_hidden, o_syn, o_memh);
}
